// HierarchicalTimeAttention_37864431681706
// MI455X (gfx1250) — matrix operand order repaired and checked by reading, not compiled, not run on hardware
//
#include <hip/hip_runtime.h>
#include <hip/hip_bf16.h>

// ---------------------------------------------------------------------------
// HierarchicalTimeAttention for MI455X (gfx1250, wave32, WMMA bf16 path)
//
//   v = x@Wv^T ; k = x@Wk^T ; r = sigmoid(x@Wr^T)        (fused, WMMA bf16)
//   lw = softmax(x@Wl^T + bl)  (D=4, VALU GEMV)
//   ds0 = state * exp(-exp(tdm))
//   o = r * (lw@ds0 + sum(lw)*k*v)                        (fused epilogue)
//   out = o@Wo^T                                          (WMMA bf16)
//   new_state = ds0 + (k*v)[t=T-1]                        (fused epilogue)
//
// A-tiles staged via GLOBAL_LOAD_ASYNC_TO_LDS_B128 (ASYNCcnt), double-buffered.
// ---------------------------------------------------------------------------

#define Bdim 4
#define Tdim 4096
#define Hdim 1024
#define Ddim 4
#define BT   (Bdim * Tdim)          // 16384 rows
#define KT   64                     // K-tile

typedef __attribute__((ext_vector_type(16))) __bf16 v16bf;
typedef __attribute__((ext_vector_type(8)))  __bf16 v8bf;
typedef __attribute__((ext_vector_type(8)))  float  v8f;
typedef int v4i_g __attribute__((vector_size(16)));   // GCC-style int4 (builtin pointee)

#if __has_builtin(__builtin_amdgcn_global_load_async_to_lds_b128)
#define HAVE_ASYNC_LDS 1
#else
#define HAVE_ASYNC_LDS 0
#endif

static __device__ __forceinline__ v16bf cat16(v8bf lo, v8bf hi) {
    return __builtin_shufflevector(lo, hi, 0,1,2,3,4,5,6,7,8,9,10,11,12,13,14,15);
}
static __device__ __forceinline__ v8bf ld8g(const unsigned short* p) {
    return *reinterpret_cast<const v8bf*>(p);
}
static __device__ __forceinline__ unsigned short f2bf(float f) {
    unsigned int u = __float_as_uint(f);
    u += 0x7FFFu + ((u >> 16) & 1u);          // RNE
    return (unsigned short)(u >> 16);
}
static __device__ __forceinline__ float sigmoidf(float x) {
    return 1.0f / (1.0f + __expf(-x));
}

// 16-byte async global->LDS copy. The AS(3) pointer is derived from the real
// generic address of the LDS element (ptrtoint -> low 32 bits -> inttoptr):
// per the gfx1250 aperture rules addr[31:0] IS the LDS byte offset, and the
// ptrtoint escapes the shared array so alias analysis keeps the ds loads.
static __device__ __forceinline__ void async_copy16(const unsigned short* g,
                                                    unsigned short* lds_dst) {
#if HAVE_ASYNC_LDS
    unsigned lds_off = (unsigned)(unsigned long long)lds_dst;   // addr[31:0]
    __builtin_amdgcn_global_load_async_to_lds_b128(
        (__attribute__((address_space(1))) v4i_g*)(unsigned long long)g,
        (__attribute__((address_space(3))) v4i_g*)(unsigned long long)lds_off,
        0, 0);
#endif
}
static __device__ __forceinline__ void wait_async0() {
#if HAVE_ASYNC_LDS
#if __has_builtin(__builtin_amdgcn_s_wait_asynccnt)
    __builtin_amdgcn_s_wait_asynccnt(0);
#else
    asm volatile("s_wait_asynccnt 0" ::: "memory");
#endif
    asm volatile("" ::: "memory");    // keep LDS reads below the wait
#endif
}

// ------------------------- fp32 -> bf16 convert ----------------------------
__global__ void cvt_bf16_kernel(const float* __restrict__ in,
                                unsigned short* __restrict__ out, long n) {
    long i = (long)blockIdx.x * blockDim.x + threadIdx.x;
    long stride = (long)gridDim.x * blockDim.x;
    for (; i < n; i += stride) out[i] = f2bf(in[i]);
}

// ------------------ decayed_state0 = state * exp(-exp(tdm)) ----------------
__global__ void ds0_kernel(const float* __restrict__ state,
                           const float* __restrict__ tdm,
                           float* __restrict__ ds0) {
    int i = blockIdx.x * blockDim.x + threadIdx.x;   // B*D*H = 16384
    if (i < Bdim * Ddim * Hdim) {
        int dh = i % (Ddim * Hdim);
        ds0[i] = state[i] * __expf(-__expf(tdm[dh]));
    }
}

// ---------------- level weights: softmax(x@Wl^T + bl), D=4 -----------------
__global__ void lw_kernel(const float* __restrict__ x,
                          const float* __restrict__ Wl,
                          const float* __restrict__ bl,
                          float* __restrict__ lw) {
    int wave = (blockIdx.x * blockDim.x + threadIdx.x) >> 5;
    int lane = threadIdx.x & 31;
    if (wave >= BT) return;
    const float* xr = x + (long)wave * Hdim;
    float a0 = 0.f, a1 = 0.f, a2 = 0.f, a3 = 0.f;
    for (int h = lane; h < Hdim; h += 32) {
        float xv = xr[h];
        a0 += xv * Wl[0 * Hdim + h];
        a1 += xv * Wl[1 * Hdim + h];
        a2 += xv * Wl[2 * Hdim + h];
        a3 += xv * Wl[3 * Hdim + h];
    }
    for (int off = 16; off > 0; off >>= 1) {
        a0 += __shfl_xor(a0, off, 32);
        a1 += __shfl_xor(a1, off, 32);
        a2 += __shfl_xor(a2, off, 32);
        a3 += __shfl_xor(a3, off, 32);
    }
    if (lane == 0) {
        float l0 = a0 + bl[0], l1 = a1 + bl[1], l2 = a2 + bl[2], l3 = a3 + bl[3];
        float m = fmaxf(fmaxf(l0, l1), fmaxf(l2, l3));
        float e0 = __expf(l0 - m), e1 = __expf(l1 - m);
        float e2 = __expf(l2 - m), e3 = __expf(l3 - m);
        float inv = 1.0f / (e0 + e1 + e2 + e3);
        float4 w = make_float4(e0 * inv, e1 * inv, e2 * inv, e3 * inv);
        reinterpret_cast<float4*>(lw)[wave] = w;
    }
}

// -------- fused v/k/r projections + attention epilogue (WMMA bf16) ---------
// grid: (H/128, BT/64), block 256 (8 waves). Wave w owns 16 cols, 64 rows.
__global__ __launch_bounds__(256)
void fused_vkr_kernel(const unsigned short* __restrict__ xh,
                      const unsigned short* __restrict__ wv,
                      const unsigned short* __restrict__ wk,
                      const unsigned short* __restrict__ wr,
                      const float* __restrict__ lw,
                      const float* __restrict__ ds0,
                      unsigned short* __restrict__ oh,
                      float* __restrict__ new_state) {
    __shared__ unsigned short ldsA[2][64 * KT];       // 2 x 8KB, double buffer

    const int lane = threadIdx.x & 31;
    const int wave = threadIdx.x >> 5;
    const int row_base = blockIdx.y * 64;
    const int ncol = blockIdx.x * 128 + wave * 16 + (lane & 15);

    const int koffB = (lane >> 4) * 16;               // B frag K offset
    const int k0A   = (lane >> 4) * 16;               // A frag K offset

    v8f accv[4], acck[4], accr[4];
    #pragma unroll
    for (int ms = 0; ms < 4; ++ms) { accv[ms] = (v8f)(0.f); acck[ms] = (v8f)(0.f); accr[ms] = (v8f)(0.f); }

    const unsigned short* bvp = wv + (long)ncol * Hdim + koffB;
    const unsigned short* bkp = wk + (long)ncol * Hdim + koffB;
    const unsigned short* brp = wr + (long)ncol * Hdim + koffB;

    // cooperative A-tile stage: thread t -> row t/4, 32 bytes at col (t&3)*16
    const int arow = threadIdx.x >> 2;
    const int acol = (threadIdx.x & 3) * 16;
    const unsigned short* agp = xh + (long)(row_base + arow) * Hdim + acol;

    auto stage = [&](int kb, int buf) {
        unsigned short* dst = &ldsA[buf][arow * KT + acol];
#if HAVE_ASYNC_LDS
        async_copy16(agp + kb, dst);
        async_copy16(agp + kb + 8, dst + 8);
#else
        *reinterpret_cast<v8bf*>(dst)     = ld8g(agp + kb);
        *reinterpret_cast<v8bf*>(dst + 8) = ld8g(agp + kb + 8);
#endif
    };

    stage(0, 0);                                      // prologue: tile 0

    for (int kt = 0; kt < Hdim / KT; ++kt) {
        const int kb = kt * KT;
        wait_async0();                                // this thread's stage landed
        __syncthreads();                              // whole tile visible; prev reads done
        if (kt + 1 < Hdim / KT) stage(kb + KT, (kt + 1) & 1);   // overlap next DMA

        if (kb + KT < Hdim) {                         // L2 warm-up for weight rows
            __builtin_prefetch(bvp + kb + KT, 0, 1);
            __builtin_prefetch(bkp + kb + KT, 0, 1);
            __builtin_prefetch(brp + kb + KT, 0, 1);
        }

        const unsigned short* lbase = &ldsA[kt & 1][0];
        #pragma unroll
        for (int kh = 0; kh < 2; ++kh) {
            const int kk = kb + kh * 32;
            v16bf Bv = cat16(ld8g(bvp + kk), ld8g(bvp + kk + 8));
            v16bf Bk = cat16(ld8g(bkp + kk), ld8g(bkp + kk + 8));
            v16bf Br = cat16(ld8g(brp + kk), ld8g(brp + kk + 8));
            #pragma unroll
            for (int ms = 0; ms < 4; ++ms) {
                const unsigned short* ap = lbase + (ms * 16 + (lane & 15)) * KT + kh * 32 + k0A;
                v16bf A = cat16(ld8g(ap), ld8g(ap + 8));
                accv[ms] = __builtin_amdgcn_wmma_f32_16x16x32_bf16(false, A, false, Bv, (short)0, accv[ms], false, false);
                acck[ms] = __builtin_amdgcn_wmma_f32_16x16x32_bf16(false, A, false, Bk, (short)0, acck[ms], false, false);
                accr[ms] = __builtin_amdgcn_wmma_f32_16x16x32_bf16(false, A, false, Br, (short)0, accr[ms], false, false);
            }
        }
    }

    // ---------------- epilogue ----------------
    const int b = row_base >> 12;                     // T = 4096 rows/batch
    float d0 = ds0[b * (Ddim * Hdim) + 0 * Hdim + ncol];
    float d1 = ds0[b * (Ddim * Hdim) + 1 * Hdim + ncol];
    float d2 = ds0[b * (Ddim * Hdim) + 2 * Hdim + ncol];
    float d3 = ds0[b * (Ddim * Hdim) + 3 * Hdim + ncol];
    const float4* lwp = reinterpret_cast<const float4*>(lw);

    #pragma unroll
    for (int ms = 0; ms < 4; ++ms) {
        #pragma unroll
        for (int i = 0; i < 8; ++i) {
            int row = row_base + ms * 16 + ((lane >> 4) << 3) + i;
            float kv = accv[ms][i] * acck[ms][i];
            float4 w = lwp[row];
            float s = w.x + w.y + w.z + w.w;
            float weighted = w.x * d0 + w.y * d1 + w.z * d2 + w.w * d3 + s * kv;
            float o = sigmoidf(accr[ms][i]) * weighted;
            oh[(long)row * Hdim + ncol] = f2bf(o);
            if ((row & (Tdim - 1)) == Tdim - 1) {     // t == T-1 -> new_state
                new_state[b * (Ddim * Hdim) + 0 * Hdim + ncol] = d0 + kv;
                new_state[b * (Ddim * Hdim) + 1 * Hdim + ncol] = d1 + kv;
                new_state[b * (Ddim * Hdim) + 2 * Hdim + ncol] = d2 + kv;
                new_state[b * (Ddim * Hdim) + 3 * Hdim + ncol] = d3 + kv;
            }
        }
    }
}

// ---------------------- final GEMM: out = o @ Wo^T -------------------------
__global__ __launch_bounds__(256)
void out_gemm_kernel(const unsigned short* __restrict__ oh,
                     const unsigned short* __restrict__ wo,
                     float* __restrict__ out) {
    __shared__ unsigned short ldsA[2][64 * KT];

    const int lane = threadIdx.x & 31;
    const int wave = threadIdx.x >> 5;
    const int row_base = blockIdx.y * 64;
    const int ncol = blockIdx.x * 128 + wave * 16 + (lane & 15);
    const int koffB = (lane >> 4) * 16;
    const int k0A   = (lane >> 4) * 16;

    v8f acc[4];
    #pragma unroll
    for (int ms = 0; ms < 4; ++ms) acc[ms] = (v8f)(0.f);

    const unsigned short* bop = wo + (long)ncol * Hdim + koffB;
    const int arow = threadIdx.x >> 2;
    const int acol = (threadIdx.x & 3) * 16;
    const unsigned short* agp = oh + (long)(row_base + arow) * Hdim + acol;

    auto stage = [&](int kb, int buf) {
        unsigned short* dst = &ldsA[buf][arow * KT + acol];
#if HAVE_ASYNC_LDS
        async_copy16(agp + kb, dst);
        async_copy16(agp + kb + 8, dst + 8);
#else
        *reinterpret_cast<v8bf*>(dst)     = ld8g(agp + kb);
        *reinterpret_cast<v8bf*>(dst + 8) = ld8g(agp + kb + 8);
#endif
    };

    stage(0, 0);

    for (int kt = 0; kt < Hdim / KT; ++kt) {
        const int kb = kt * KT;
        wait_async0();
        __syncthreads();
        if (kt + 1 < Hdim / KT) stage(kb + KT, (kt + 1) & 1);
        if (kb + KT < Hdim) __builtin_prefetch(bop + kb + KT, 0, 1);

        const unsigned short* lbase = &ldsA[kt & 1][0];
        #pragma unroll
        for (int kh = 0; kh < 2; ++kh) {
            const int kk = kb + kh * 32;
            v16bf Bo = cat16(ld8g(bop + kk), ld8g(bop + kk + 8));
            #pragma unroll
            for (int ms = 0; ms < 4; ++ms) {
                const unsigned short* ap = lbase + (ms * 16 + (lane & 15)) * KT + kh * 32 + k0A;
                v16bf A = cat16(ld8g(ap), ld8g(ap + 8));
                acc[ms] = __builtin_amdgcn_wmma_f32_16x16x32_bf16(false, A, false, Bo, (short)0, acc[ms], false, false);
            }
        }
    }

    #pragma unroll
    for (int ms = 0; ms < 4; ++ms) {
        #pragma unroll
        for (int i = 0; i < 8; ++i) {
            int row = row_base + ms * 16 + ((lane >> 4) << 3) + i;
            out[(long)row * Hdim + ncol] = acc[ms][i];
        }
    }
}

// ---------------------------------------------------------------------------
extern "C" void kernel_launch(void* const* d_in, const int* in_sizes, int n_in,
                              void* d_out, int out_size, void* d_ws, size_t ws_size,
                              hipStream_t stream) {
    (void)in_sizes; (void)n_in; (void)out_size; (void)ws_size;

    const float* x     = (const float*)d_in[0];   // [B,T,H]
    const float* state = (const float*)d_in[1];   // [B,D,H]
    const float* tdm   = (const float*)d_in[2];   // [D,H]
    const float* Wl    = (const float*)d_in[3];   // [D,H]
    const float* bl    = (const float*)d_in[4];   // [D]
    const float* Wv    = (const float*)d_in[5];   // [H,H]
    const float* Wk    = (const float*)d_in[6];
    const float* Wr    = (const float*)d_in[7];
    const float* Wo    = (const float*)d_in[8];

    float* out       = (float*)d_out;                      // [B,T,H]
    float* new_state = out + (long)BT * Hdim;              // [B,D,H]

    // workspace carving
    char* ws = (char*)d_ws;
    unsigned short* xh  = (unsigned short*)ws;  ws += (long)BT * Hdim * 2;       // 32 MB
    unsigned short* wvh = (unsigned short*)ws;  ws += (long)Hdim * Hdim * 2;     // 2 MB
    unsigned short* wkh = (unsigned short*)ws;  ws += (long)Hdim * Hdim * 2;
    unsigned short* wrh = (unsigned short*)ws;  ws += (long)Hdim * Hdim * 2;
    unsigned short* woh = (unsigned short*)ws;  ws += (long)Hdim * Hdim * 2;
    unsigned short* oh  = (unsigned short*)ws;  ws += (long)BT * Hdim * 2;       // 32 MB
    float* lw  = (float*)ws;                    ws += (long)BT * 4 * 4;          // 256 KB
    float* ds0 = (float*)ws;                    ws += (long)Bdim * Ddim * Hdim * 4;

    // 1) converts
    cvt_bf16_kernel<<<8192, 256, 0, stream>>>(x,  xh,  (long)BT * Hdim);
    cvt_bf16_kernel<<<2048, 256, 0, stream>>>(Wv, wvh, (long)Hdim * Hdim);
    cvt_bf16_kernel<<<2048, 256, 0, stream>>>(Wk, wkh, (long)Hdim * Hdim);
    cvt_bf16_kernel<<<2048, 256, 0, stream>>>(Wr, wrh, (long)Hdim * Hdim);
    cvt_bf16_kernel<<<2048, 256, 0, stream>>>(Wo, woh, (long)Hdim * Hdim);

    // 2) decayed state
    ds0_kernel<<<(Bdim * Ddim * Hdim + 255) / 256, 256, 0, stream>>>(state, tdm, ds0);

    // 3) level weights
    lw_kernel<<<BT / 8, 256, 0, stream>>>(x, Wl, bl, lw);

    // 4) fused v/k/r + attention epilogue + new_state
    fused_vkr_kernel<<<dim3(Hdim / 128, BT / 64), 256, 0, stream>>>(
        xh, wvh, wkh, wrh, lw, ds0, oh, new_state);

    // 5) output projection
    out_gemm_kernel<<<dim3(Hdim / 128, BT / 64), 256, 0, stream>>>(oh, woh, out);
}
